// MultiHeadCrossAttention_75041668596308
// MI455X (gfx1250) — matrix operand order repaired and checked by reading, not compiled, not run on hardware
//
#include <hip/hip_runtime.h>
#include <hip/hip_bf16.h>

typedef __attribute__((ext_vector_type(16))) _Float16 v16h;
typedef __attribute__((ext_vector_type(8)))  _Float16 v8h;
typedef __attribute__((ext_vector_type(4)))  _Float16 v4h;
typedef __attribute__((ext_vector_type(8)))  float    v8f;

#define DIM   1024
#define HEADS 16
#define HD    64
#define SEQ   2048
#define BATCH 4

// ---------------------------------------------------------------------------
// fp32 -> fp16 conversion (vectorized 4-wide)
// ---------------------------------------------------------------------------
__global__ __launch_bounds__(256) void cvt_kernel(const float* __restrict__ in,
                                                  _Float16* __restrict__ out,
                                                  int n4) {
  int i = blockIdx.x * 256 + threadIdx.x;
  if (i < n4) {
    float4 v = ((const float4*)in)[i];
    v4h h;
    h[0] = (_Float16)v.x; h[1] = (_Float16)v.y;
    h[2] = (_Float16)v.z; h[3] = (_Float16)v.w;
    ((v4h*)out)[i] = h;
  }
}

// ---------------------------------------------------------------------------
// Projection GEMM: OUT = X @ W^T + bias  (X: 8192x1024 f16, W: 1024x1024 f16)
// One wave -> 16 rows x 64 cols. Stores per-head layout:
//   vtrans==0: OUT[(b,h,t,hd)]   (for Q and K)
//   vtrans==1: OUT[(b,h,hd,s)]   (for V, transposed so attn B-frags are rows)
// ---------------------------------------------------------------------------
__global__ __launch_bounds__(256) void proj_kernel(const _Float16* __restrict__ X,
                                                   const _Float16* __restrict__ W,
                                                   const float* __restrict__ bias,
                                                   _Float16* __restrict__ OUT,
                                                   int vtrans) {
  const int lane = threadIdx.x & 31;
  const int wv   = threadIdx.x >> 5;
  const int lm   = lane & 15;
  const bool lo  = lane < 16;

  const int task    = blockIdx.x * 8 + wv;     // 8192 wave tasks
  const int rowtile = task >> 4;               // 512 row tiles of 16
  const int coltile = task & 15;               // 16 col tiles of 64
  const int row0    = rowtile << 4;
  const int n0      = coltile << 6;

  v8f acc[4] = {};

  const _Float16* arow = X + (size_t)(row0 + lm) * DIM;
  const int ka = lo ? 0 : 16;    // A-frag halves 0..7  -> K {0..7 | 16..23}
  const int kb = lo ? 8 : 24;    // A-frag halves 8..15 -> K {8..15 | 24..31}

  for (int kk = 0; kk < DIM; kk += 32) {
    v8h a0 = *(const v8h*)(arow + kk + ka);
    v8h a1 = *(const v8h*)(arow + kk + kb);
    v16h a;
#pragma unroll
    for (int i = 0; i < 8; ++i) { a[i] = a0[i]; a[8 + i] = a1[i]; }

    const int kw = kk + (lo ? 0 : 16);   // B-frag: lanes 0-15 K=kk..kk+15, 16-31 K=kk+16..kk+31
#pragma unroll
    for (int ct = 0; ct < 4; ++ct) {
      v16h b = *(const v16h*)(W + (size_t)(n0 + ct * 16 + lm) * DIM + kw);
      acc[ct] = __builtin_amdgcn_wmma_f32_16x16x32_f16(
          false, a, false, b, (short)0, acc[ct], false, false);
    }
  }

#pragma unroll
  for (int ct = 0; ct < 4; ++ct) {
    const int n  = n0 + ct * 16 + lm;       // output column (this lane's N)
    const float bn = bias[n];
    const int h  = n >> 6;
    const int hd = n & 63;
#pragma unroll
    for (int r = 0; r < 8; ++r) {
      const int gr = row0 + r + (lo ? 0 : 8);   // global row (C layout)
      const int b2 = gr >> 11;
      const int tt = gr & 2047;
      const float val = acc[ct][r] + bn;
      const size_t idx = vtrans
          ? ((((size_t)b2 * HEADS + h) * HD + hd) * SEQ + tt)
          : ((((size_t)b2 * HEADS + h) * SEQ + tt) * HD + hd);
      OUT[idx] = (_Float16)val;
    }
  }
}

// ---------------------------------------------------------------------------
// Flash attention: one wave owns a 16-row Q tile, loops s in 64-wide tiles.
// ---------------------------------------------------------------------------
__global__ __launch_bounds__(256) void attn_kernel(const _Float16* __restrict__ QP,
                                                   const _Float16* __restrict__ KP,
                                                   const _Float16* __restrict__ VPT,
                                                   const int* __restrict__ mask,
                                                   float* __restrict__ out) {
  __shared__ __align__(32) _Float16 plds[8][16][64];   // per-wave P tile (16KB)

  const int lane = threadIdx.x & 31;
  const int wv   = threadIdx.x >> 5;
  const int lm   = lane & 15;
  const bool lo  = lane < 16;

  const int bh = blockIdx.x >> 4;          // 64 (b,h) pairs
  const int b  = bh >> 4;
  const int h  = bh & 15;
  const int q0 = ((blockIdx.x & 15) << 7) + (wv << 4);

  // Q A-fragments (two K-chunks over head_dim=64), kept in registers
  const _Float16* qrow = QP + ((size_t)bh * SEQ + q0 + lm) * HD;
  v16h aq[2];
#pragma unroll
  for (int ch = 0; ch < 2; ++ch) {
    v8h a0 = *(const v8h*)(qrow + 32 * ch + (lo ? 0 : 16));
    v8h a1 = *(const v8h*)(qrow + 32 * ch + (lo ? 8 : 24));
#pragma unroll
    for (int i = 0; i < 8; ++i) { aq[ch][i] = a0[i]; aq[ch][8 + i] = a1[i]; }
  }

  v8f o[4] = {};
  float mrow[8], lrow[8];
#pragma unroll
  for (int r = 0; r < 8; ++r) { mrow[r] = -3.0e38f; lrow[r] = 0.0f; }

  const int* mbase = mask + ((size_t)b * SEQ + q0) * SEQ;

  for (int s0 = 0; s0 < SEQ; s0 += 64) {
    // ---- S = Q @ K^T over this 16x64 tile ----
    v8f sc[4] = {};
#pragma unroll
    for (int ct = 0; ct < 4; ++ct) {
      const _Float16* krow =
          KP + ((size_t)bh * SEQ + s0 + ct * 16 + lm) * HD + (lo ? 0 : 16);
#pragma unroll
      for (int ch = 0; ch < 2; ++ch) {
        v16h bf = *(const v16h*)(krow + 32 * ch);
        sc[ct] = __builtin_amdgcn_wmma_f32_16x16x32_f16(
            false, aq[ch], false, bf, (short)0, sc[ct], false, false);
      }
    }

    // ---- scale by 1/sqrt(DIM)=1/32 and apply (mask==1 -> -inf) ----
#pragma unroll
    for (int ct = 0; ct < 4; ++ct) {
#pragma unroll
      for (int r = 0; r < 8; ++r) {
        const int rr = r + (lo ? 0 : 8);
        const int cc = s0 + ct * 16 + lm;
        const int mk = mbase[(size_t)rr * SEQ + cc];
        const float v = sc[ct][r] * 0.03125f;
        sc[ct][r] = (mk == 1) ? -3.0e38f : v;
      }
    }

    // ---- online softmax (row reductions across 16-lane half-waves) ----
#pragma unroll
    for (int r = 0; r < 8; ++r) {
      float tv = fmaxf(fmaxf(sc[0][r], sc[1][r]), fmaxf(sc[2][r], sc[3][r]));
#pragma unroll
      for (int off = 8; off > 0; off >>= 1) tv = fmaxf(tv, __shfl_xor(tv, off));
      const float nm    = fmaxf(mrow[r], tv);
      const float alpha = __expf(mrow[r] - nm);
      mrow[r] = nm;
      lrow[r] *= alpha;
#pragma unroll
      for (int ct = 0; ct < 4; ++ct) o[ct][r] *= alpha;
      float s = 0.0f;
#pragma unroll
      for (int ct = 0; ct < 4; ++ct) {
        const float p = __expf(sc[ct][r] - nm);
        sc[ct][r] = p;
        s += p;
      }
#pragma unroll
      for (int off = 8; off > 0; off >>= 1) s += __shfl_xor(s, off);
      lrow[r] += s;
    }

    // ---- P (C layout) -> LDS -> P A-fragments ----
#pragma unroll
    for (int ct = 0; ct < 4; ++ct)
#pragma unroll
      for (int r = 0; r < 8; ++r)
        plds[wv][r + (lo ? 0 : 8)][ct * 16 + lm] = (_Float16)sc[ct][r];

    asm volatile("s_wait_dscnt 0" ::: "memory");  // wave-local store->load order

    v16h pa[2];
    const _Float16* prow = &plds[wv][lm][0];
#pragma unroll
    for (int ch = 0; ch < 2; ++ch) {
      v8h a0 = *(const v8h*)(prow + 32 * ch + (lo ? 0 : 16));
      v8h a1 = *(const v8h*)(prow + 32 * ch + (lo ? 8 : 24));
#pragma unroll
      for (int i = 0; i < 8; ++i) { pa[ch][i] = a0[i]; pa[ch][8 + i] = a1[i]; }
    }

    // ---- O += P @ V (VPT is (b,h,hd,s): rows are contiguous in s) ----
#pragma unroll
    for (int ct = 0; ct < 4; ++ct) {
      const _Float16* vrow =
          VPT + ((size_t)bh * HD + ct * 16 + lm) * SEQ + s0 + (lo ? 0 : 16);
#pragma unroll
      for (int ch = 0; ch < 2; ++ch) {
        v16h bf = *(const v16h*)(vrow + 32 * ch);
        o[ct] = __builtin_amdgcn_wmma_f32_16x16x32_f16(
            false, pa[ch], false, bf, (short)0, o[ct], false, false);
      }
    }
  }

  // ---- epilogue: out[b, t, h*64 + hd] = o / l ----
#pragma unroll
  for (int ct = 0; ct < 4; ++ct) {
#pragma unroll
    for (int r = 0; r < 8; ++r) {
      const int rr  = q0 + r + (lo ? 0 : 8);
      const int col = h * 64 + ct * 16 + lm;
      out[((size_t)b * SEQ + rr) * DIM + col] = o[ct][r] / lrow[r];
    }
  }
}

// ---------------------------------------------------------------------------
extern "C" void kernel_launch(void* const* d_in, const int* in_sizes, int n_in,
                              void* d_out, int out_size, void* d_ws, size_t ws_size,
                              hipStream_t stream) {
  const float* q    = (const float*)d_in[0];
  const float* k    = (const float*)d_in[1];
  const float* v    = (const float*)d_in[2];
  const int*   msk  = (const int*)  d_in[3];
  const float* Wq   = (const float*)d_in[4];
  const float* bq   = (const float*)d_in[5];
  const float* Wk   = (const float*)d_in[6];
  const float* bk   = (const float*)d_in[7];
  const float* Wv   = (const float*)d_in[8];
  const float* bv   = (const float*)d_in[9];
  float* out = (float*)d_out;

  const size_t NQ = (size_t)BATCH * SEQ * DIM;   // 8,388,608
  const size_t NW = (size_t)DIM * DIM;           // 1,048,576

  _Float16* ws   = (_Float16*)d_ws;
  _Float16* q16  = ws;
  _Float16* k16  = q16 + NQ;
  _Float16* v16  = k16 + NQ;
  _Float16* Wq16 = v16 + NQ;
  _Float16* Wk16 = Wq16 + NW;
  _Float16* Wv16 = Wk16 + NW;
  _Float16* QP   = Wv16 + NW;
  _Float16* KP   = QP + NQ;
  _Float16* VPT  = KP + NQ;

  // fp32 -> fp16
  cvt_kernel<<<(int)(NQ / 4 / 256), 256, 0, stream>>>(q, q16, (int)(NQ / 4));
  cvt_kernel<<<(int)(NQ / 4 / 256), 256, 0, stream>>>(k, k16, (int)(NQ / 4));
  cvt_kernel<<<(int)(NQ / 4 / 256), 256, 0, stream>>>(v, v16, (int)(NQ / 4));
  cvt_kernel<<<(int)(NW / 4 / 256), 256, 0, stream>>>(Wq, Wq16, (int)(NW / 4));
  cvt_kernel<<<(int)(NW / 4 / 256), 256, 0, stream>>>(Wk, Wk16, (int)(NW / 4));
  cvt_kernel<<<(int)(NW / 4 / 256), 256, 0, stream>>>(Wv, Wv16, (int)(NW / 4));

  // projections (1024 blocks x 8 waves = 8192 wave tiles of 16x64)
  proj_kernel<<<1024, 256, 0, stream>>>(q16, Wq16, bq, QP, 0);
  proj_kernel<<<1024, 256, 0, stream>>>(k16, Wk16, bk, KP, 0);
  proj_kernel<<<1024, 256, 0, stream>>>(v16, Wv16, bv, VPT, 1);

  // flash attention: 64 (b,h) pairs x 16 q-blocks of 128 rows
  attn_kernel<<<1024, 256, 0, stream>>>(QP, KP, VPT, msk, out);
}
